// SingleScaleVolSDF_65541200937475
// MI455X (gfx1250) — matrix operand order repaired and checked by reading, not compiled, not run on hardware
//
#include <hip/hip_runtime.h>
#include <hip/hip_bf16.h>
#include <math.h>

typedef _Float16 h16 __attribute__((ext_vector_type(16)));
typedef float    f8  __attribute__((ext_vector_type(8)));

#define NRAYS 1024
#define NSAMP 500
#define GR    128
#define GCH   13          // 1 sdf + 12 latent
#define HID   128
#define NPTS  (NRAYS * NSAMP)

// ---------------------------------------------------------------------------
// Prep: fold W_feat into W1 (K: 73 -> 21, pad to 32) and pre-swizzle the
// folded W1' and zero-padded W2 into WMMA B-fragment lane order.
// B 32x16 f16 layout: lane L holds column N = L%16, halves j=0..7 are
// K = (L<16 ? 0 : 8) + j, halves 8..15 same +16  (plus 32*chunk for layer 2).
// ---------------------------------------------------------------------------
__global__ void prep_kernel(const float* __restrict__ W_feat,
                            const float* __restrict__ b_feat,
                            const float* __restrict__ W1,
                            const float* __restrict__ b1,
                            const float* __restrict__ W2,
                            _Float16* __restrict__ W1frag,   // [8 ntiles][32 lanes][16]
                            _Float16* __restrict__ W2frag,   // [4 kchunks][32 lanes][16]
                            float* __restrict__ b1p)         // [128]
{
    int t = threadIdx.x;               // 256 threads, 1 block
    {
        int n = t >> 5, L = t & 31;
        int nc = (n << 4) + (L & 15);
        int kbase = (L < 16) ? 0 : 8;
        #pragma unroll
        for (int j = 0; j < 16; ++j) {
            int k = kbase + (j < 8 ? j : j + 8);
            float v = 0.0f;
            if (k < 9) {
                v = W1[k * HID + nc];
            } else if (k < 21) {
                int kk = k - 9;
                float acc = 0.0f;
                for (int jj = 0; jj < 64; ++jj)
                    acc += W_feat[kk * 64 + jj] * W1[(9 + jj) * HID + nc];
                v = acc;
            }
            W1frag[(n * 32 + L) * 16 + j] = (_Float16)v;
        }
    }
    if (t < 128) {                      // folded bias: b1' = b1 + b_feat @ W1[9:73]
        float acc = b1[t];
        for (int j = 0; j < 64; ++j) acc += b_feat[j] * W1[(9 + j) * HID + t];
        b1p[t] = acc;
    }
    if (t < 128) {                      // W2 (128x3) zero-padded to 128x16, fragment order
        int c = t >> 5, L = t & 31;
        int ncol = L & 15;
        int kbase = c * 32 + ((L < 16) ? 0 : 8);
        #pragma unroll
        for (int j = 0; j < 16; ++j) {
            int k = kbase + (j < 8 ? j : j + 8);
            float v = (ncol < 3) ? W2[k * 3 + ncol] : 0.0f;
            W2frag[(c * 32 + L) * 16 + j] = (_Float16)v;
        }
    }
}

// ---------------------------------------------------------------------------
// Main kernel: 128 threads = 4 waves; each wave owns 32 points (two M-tiles).
// Phase 1 (ALL 32 lanes): trilerp gather (L2-resident grid) + SDF gradient
//                         -> h'(21, padded 32) into LDS, one point per lane.
// Phase 2 (per 16-row tile): layer1 = 8x v_wmma_f32_16x16x32_f16 (K=32),
//                            relu -> LDS, layer2 = 4x wmma (K=128),
//                            sigmoid+mask -> rgb.
// ---------------------------------------------------------------------------
__global__ void __launch_bounds__(128)
mlp_kernel(const float* __restrict__ rays_o, const float* __restrict__ rays_d,
           const float* __restrict__ viewdirs, const float* __restrict__ grid,
           const float* __restrict__ b2,
           const _Float16* __restrict__ W1frag, const _Float16* __restrict__ W2frag,
           const float* __restrict__ b1p,
           float* __restrict__ sdf_out, float* __restrict__ rgb_out)
{
    __shared__ _Float16 sh_h[4][32][32];     // per-wave h' (32 points, K padded to 32)
    __shared__ _Float16 sh_hid[4][32][HID];  // per-wave hidden activations
    __shared__ int      sh_m[4][32];         // per-point out-of-box mask

    const int tid = threadIdx.x;
    const int w   = tid >> 5;
    const int L   = tid & 31;
    const int wave_base = blockIdx.x * 128 + w * 32;   // 32 points per wave

    // ---------------- Phase 1: per-point geometry + grid gather (all lanes) --
    {
        int p   = wave_base + L;
        int ray = p / NSAMP;
        int s   = p - ray * NSAMP;
        float z = 0.5f + 3.0f * (float)s / 499.0f;

        float ox = rays_o[ray * 3 + 0], oy = rays_o[ray * 3 + 1], oz = rays_o[ray * 3 + 2];
        float dx = rays_d[ray * 3 + 0], dy = rays_d[ray * 3 + 1], dzr = rays_d[ray * 3 + 2];
        float vx = viewdirs[ray * 3 + 0], vy = viewdirs[ray * 3 + 1], vz = viewdirs[ray * 3 + 2];

        float px = ox + dx * z, py = oy + dy * z, pz = oz + dzr * z;
        int mask = (fabsf(px) > 1.0f) || (fabsf(py) > 1.0f) || (fabsf(pz) > 1.0f);

        float ux = fminf(fmaxf((px + 1.0f) * 63.5f, 0.0f), 127.0f);
        float uy = fminf(fmaxf((py + 1.0f) * 63.5f, 0.0f), 127.0f);
        float uz = fminf(fmaxf((pz + 1.0f) * 63.5f, 0.0f), 127.0f);
        int ix = min((int)floorf(ux), 126);
        int iy = min((int)floorf(uy), 126);
        int iz = min((int)floorf(uz), 126);
        float fx = ux - (float)ix, fy = uy - (float)iy, fz = uz - (float)iz;

        float acc[GCH];
        #pragma unroll
        for (int c = 0; c < GCH; ++c) acc[c] = 0.0f;
        float cs[8];

        #pragma unroll
        for (int d0 = 0; d0 < 2; ++d0)
        #pragma unroll
        for (int d1 = 0; d1 < 2; ++d1)
        #pragma unroll
        for (int d2 = 0; d2 < 2; ++d2) {
            float wgt = (d0 ? fx : 1.0f - fx) * (d1 ? fy : 1.0f - fy) * (d2 ? fz : 1.0f - fz);
            const float* g = grid + (size_t)((((ix + d0) * GR + (iy + d1)) * GR + (iz + d2)) * GCH);
            float v0 = g[0];
            cs[d0 * 4 + d1 * 2 + d2] = v0;
            acc[0] += wgt * v0;
            #pragma unroll
            for (int c = 1; c < GCH; ++c) acc[c] += wgt * g[c];
        }

        float wx0 = 1.0f - fx, wx1 = fx, wy0 = 1.0f - fy, wy1 = fy, wz0 = 1.0f - fz, wz1 = fz;
        float gx = 63.5f * (wy0 * wz0 * (cs[4] - cs[0]) + wy0 * wz1 * (cs[5] - cs[1]) +
                            wy1 * wz0 * (cs[6] - cs[2]) + wy1 * wz1 * (cs[7] - cs[3]));
        float gy = 63.5f * (wx0 * wz0 * (cs[2] - cs[0]) + wx0 * wz1 * (cs[3] - cs[1]) +
                            wx1 * wz0 * (cs[6] - cs[4]) + wx1 * wz1 * (cs[7] - cs[5]));
        float gz = 63.5f * (wx0 * wy0 * (cs[1] - cs[0]) + wx0 * wy1 * (cs[3] - cs[2]) +
                            wx1 * wy0 * (cs[5] - cs[4]) + wx1 * wy1 * (cs[7] - cs[6]));

        sdf_out[p] = mask ? 100.0f : acc[0];
        sh_m[w][L] = mask;

        _Float16* hr = &sh_h[w][L][0];
        hr[0] = (_Float16)px; hr[1] = (_Float16)py; hr[2] = (_Float16)pz;
        hr[3] = (_Float16)gx; hr[4] = (_Float16)gy; hr[5] = (_Float16)gz;
        hr[6] = (_Float16)vx; hr[7] = (_Float16)vy; hr[8] = (_Float16)vz;
        #pragma unroll
        for (int c = 0; c < 12; ++c) hr[9 + c] = (_Float16)acc[1 + c];
        #pragma unroll
        for (int k = 21; k < 32; ++k) hr[k] = (_Float16)0.0f;
    }
    __syncthreads();

    // ---------------- Phase 2: two 16-point M-tiles per wave -----------------
    // A 16x32 f16 layout: lane L -> row M=L%16; halves 0..7 = K (L<16?0:8)+j,
    // halves 8..15 = same +16.
    const int M     = L & 15;
    const int abase = (L < 16) ? 0 : 8;
    const int col   = L & 15;
    const float bias2 = (col < 3) ? b2[col] : 0.0f;

    #pragma unroll 1                                   // keep one tile in flight
    for (int t = 0; t < 2; ++t) {
        const int row = t * 16;

        h16 a;
        #pragma unroll
        for (int j = 0; j < 8; ++j) {
            a[j]     = sh_h[w][row + M][abase + j];
            a[8 + j] = sh_h[w][row + M][abase + 16 + j];
        }

        // -------- layer 1: 8 N-tiles of HID=128 --------
        #pragma unroll
        for (int n = 0; n < 8; ++n) {
            h16 bfr = *(const h16*)(W1frag + (n * 32 + L) * 16);
            float bv = b1p[n * 16 + col];
            f8 c;
            #pragma unroll
            for (int r = 0; r < 8; ++r) c[r] = bv;
            c = __builtin_amdgcn_wmma_f32_16x16x32_f16(false, a, false, bfr,
                                                       (short)0, c, false, false);
            #pragma unroll
            for (int r = 0; r < 8; ++r) {
                float v = fmaxf(c[r], 0.0f);                       // relu
                int m = (L < 16) ? r : (r + 8);                    // C layout row
                sh_hid[w][row + m][n * 16 + col] = (_Float16)v;
            }
        }
        __syncthreads();

        // -------- layer 2: K=128 in 4 chunks, N padded 3->16 --------
        f8 c2;
        #pragma unroll
        for (int r = 0; r < 8; ++r) c2[r] = bias2;

        #pragma unroll
        for (int cc = 0; cc < 4; ++cc) {
            h16 a2;
            int kb = cc * 32 + abase;
            #pragma unroll
            for (int j = 0; j < 8; ++j) {
                a2[j]     = sh_hid[w][row + M][kb + j];
                a2[8 + j] = sh_hid[w][row + M][kb + 16 + j];
            }
            h16 b2f = *(const h16*)(W2frag + (cc * 32 + L) * 16);
            c2 = __builtin_amdgcn_wmma_f32_16x16x32_f16(false, a2, false, b2f,
                                                        (short)0, c2, false, false);
        }

        if (col < 3) {
            #pragma unroll
            for (int r = 0; r < 8; ++r) {
                int m = (L < 16) ? r : (r + 8);
                int p = wave_base + row + m;
                float v = sh_m[w][row + m] ? 0.0f : (1.0f / (1.0f + expf(-c2[r])));
                rgb_out[p * 3 + col] = v;
            }
        }
        __syncthreads();
    }
}

// ---------------------------------------------------------------------------
// Per-ray alpha compositing: serial 500-step scan, one thread per ray.
// ---------------------------------------------------------------------------
__global__ void composite_kernel(const float* __restrict__ sdf,
                                 const float* __restrict__ rgbp,
                                 const float* __restrict__ beta_ptr,
                                 float* __restrict__ out)
{
    int r = blockIdx.x * blockDim.x + threadIdx.x;
    if (r >= NRAYS) return;
    float b_eff = fabsf(beta_ptr[0]) + 1e-4f;
    float inv_b = 1.0f / b_eff;
    const float dz = 3.0f / 499.0f;

    float T = 1.0f, r0 = 0.0f, r1 = 0.0f, r2 = 0.0f, dep = 0.0f;
    const float* srow = sdf + (size_t)r * NSAMP;
    const float* crow = rgbp + (size_t)r * NSAMP * 3;

    for (int s = 0; s < NSAMP; ++s) {
        float sv  = srow[s];
        float sgn = (sv > 0.0f) ? 1.0f : ((sv < 0.0f) ? -1.0f : 0.0f);
        float dens = inv_b * (0.5f + 0.5f * sgn * expm1f(-fabsf(sv) * inv_b));
        float fe   = dz * dens;
        float efe  = expf(-fe);
        float wgt  = (1.0f - efe) * T;
        float z    = 0.5f + 3.0f * (float)s / 499.0f;   // |o - pt| = z (unit dirs)
        r0 += wgt * crow[s * 3 + 0];
        r1 += wgt * crow[s * 3 + 1];
        r2 += wgt * crow[s * 3 + 2];
        dep += wgt * z;
        T *= efe;
    }
    out[r * 3 + 0] = r0;
    out[r * 3 + 1] = r1;
    out[r * 3 + 2] = r2;
    out[3 * NRAYS + r] = dep;
}

// ---------------------------------------------------------------------------
extern "C" void kernel_launch(void* const* d_in, const int* in_sizes, int n_in,
                              void* d_out, int out_size, void* d_ws, size_t ws_size,
                              hipStream_t stream)
{
    const float* rays_o   = (const float*)d_in[0];
    const float* rays_d   = (const float*)d_in[1];
    const float* viewdirs = (const float*)d_in[2];
    const float* grid     = (const float*)d_in[3];
    const float* W_feat   = (const float*)d_in[4];
    const float* b_feat   = (const float*)d_in[5];
    const float* W1       = (const float*)d_in[6];
    const float* b1       = (const float*)d_in[7];
    const float* W2       = (const float*)d_in[8];
    const float* b2       = (const float*)d_in[9];
    const float* beta     = (const float*)d_in[10];

    char* ws = (char*)d_ws;
    _Float16* W1frag = (_Float16*)ws;  ws += 8 * 32 * 16 * sizeof(_Float16);  // 8 KB
    _Float16* W2frag = (_Float16*)ws;  ws += 4 * 32 * 16 * sizeof(_Float16);  // 4 KB
    float*    b1p    = (float*)ws;     ws += 128 * sizeof(float);
    float*    sdf    = (float*)ws;     ws += (size_t)NPTS * sizeof(float);        // 2.0 MB
    float*    rgbp   = (float*)ws;     ws += (size_t)NPTS * 3 * sizeof(float);    // 6.1 MB

    prep_kernel<<<1, 256, 0, stream>>>(W_feat, b_feat, W1, b1, W2, W1frag, W2frag, b1p);
    mlp_kernel<<<NPTS / 128, 128, 0, stream>>>(rays_o, rays_d, viewdirs, grid, b2,
                                               W1frag, W2frag, b1p, sdf, rgbp);
    composite_kernel<<<(NRAYS + 255) / 256, 256, 0, stream>>>(sdf, rgbp, beta, (float*)d_out);
}
